// Transformer_67181878444047
// MI455X (gfx1250) — matrix operand order repaired and checked by reading, not compiled, not run on hardware
//
#include <hip/hip_runtime.h>

// ---------------------------------------------------------------------------
// Transformer forward for MI455X (gfx1250, wave32, WMMA f32_16x16x32_f16).
// All matmuls run on v_wmma_f32_16x16x32_f16; f32 master data is converted to
// f16 inside the fragment loads (everything is L2-resident on a 192MB L2).
// K-loop is software-pipelined with double-buffered fragments to kill the
// WMMA->VALU hazard NOPs; N-subtile count is a template param so no runtime
// guards remain in the hot loop.
// ---------------------------------------------------------------------------

typedef __attribute__((ext_vector_type(16))) _Float16 v16h;
typedef __attribute__((ext_vector_type(8)))  _Float16 v8h;
typedef __attribute__((ext_vector_type(8)))  float    v8f;

#define TN  2       // batch
#define TS  1024    // seq len
#define TE  256     // embed
#define TH  8       // heads
#define TD  32      // head dim
#define TL  6       // layers
#define TFE 4       // ffn expansion
#define TNTOK (TN*TS)
#define TNH   (TN*TH)

// ---------------- fragment loads (A: 16x32 f16 frag, B: 32x16 f16 frag) -----

__device__ inline v16h load_afrag(const _Float16* p) {
  v8h lo = *(const v8h*)p;          // K chunk 0 (8 halves)
  v8h hi = *(const v8h*)(p + 8);    // K chunk +8
  v16h r;
#pragma unroll
  for (int i = 0; i < 8; ++i) { r[i] = lo[i]; r[i + 8] = hi[i]; }
  return r;
}
__device__ inline v16h load_afrag(const float* p) {
  v8f lo = *(const v8f*)p;
  v8f hi = *(const v8f*)(p + 8);
  v16h r;
#pragma unroll
  for (int i = 0; i < 8; ++i) { r[i] = (_Float16)lo[i]; r[i + 8] = (_Float16)hi[i]; }
  return r;
}
__device__ inline v16h load_bfrag(const _Float16* p) {
  return *(const v16h*)p;           // 16 contiguous halves (32B)
}
__device__ inline v16h load_bfrag(const float* p) {
  v8f lo = *(const v8f*)p;
  v8f hi = *(const v8f*)(p + 8);
  v16h r;
#pragma unroll
  for (int i = 0; i < 8; ++i) { r[i] = (_Float16)lo[i]; r[i + 8] = (_Float16)hi[i]; }
  return r;
}

__device__ inline v8f wmma_step(v16h a, v16h b, v8f c) {
  return __builtin_amdgcn_wmma_f32_16x16x32_f16(false, a, false, b, (short)0,
                                                c, false, false);
}

// ---------------- generic batched GEMM:  C = A @ B^T (+bias)(+relu) --------
// A: [M,K] rows stride lda.  B: [Nc,K] rows stride ldb.  C(m,n) at
// m*rsC + n*csC.  Two-level batch: b -> (b % bi) inner, (b / bi) outer.
// NT = number of 16-wide N subtiles per wave; callers guarantee Nc % (16*NT)
// == 0 and M % 128 == 0 for the grids used here.

struct GArgs {
  const void* A; const void* B; const float* bias; void* C;
  long long oA, sAi, sAo;
  long long oB, sBi, sBo;
  long long oC, sCi, sCo;
  int lda, ldb, rsC, csC;
  int M, Nc, K, bi, nb;
};

template <typename TA, typename TB, typename TC, bool BIAS, bool RELU, int NT>
__global__ __launch_bounds__(128) void gemm_wmma(GArgs g) {
  const int lane = threadIdx.x & 31;
  const int wave = threadIdx.x >> 5;
  const int b    = blockIdx.z;
  const int bin  = b % g.bi;
  const int bout = b / g.bi;

  const TA* A = (const TA*)g.A + g.oA + (long long)bin * g.sAi + (long long)bout * g.sAo;
  const TB* B = (const TB*)g.B + g.oB + (long long)bin * g.sBi + (long long)bout * g.sBo;
  TC*       C = (TC*)g.C       + g.oC + (long long)bin * g.sCi + (long long)bout * g.sCo;

  const int m0 = blockIdx.y * 128 + wave * 32;
  if (m0 >= g.M) return;
  const int n0 = blockIdx.x * (16 * NT);

  const int arow = lane & 15, kc = (lane >> 4) * 16;   // A frag lane mapping
  const int bcol = lane & 15, kb = (lane >> 4) * 16;   // B frag lane mapping

  const TA* ap0 = A + (long long)(m0 + arow) * g.lda + kc;
  const TA* ap1 = ap0 + 16ll * g.lda;
  const TB* bp[NT];
#pragma unroll
  for (int ni = 0; ni < NT; ++ni)
    bp[ni] = B + (long long)(n0 + ni * 16 + bcol) * g.ldb + kb;

  v8f acc[2][NT];
#pragma unroll
  for (int mi = 0; mi < 2; ++mi)
#pragma unroll
    for (int ni = 0; ni < NT; ++ni)
#pragma unroll
      for (int r = 0; r < 8; ++r) acc[mi][ni][r] = 0.0f;

  // ---- software-pipelined K loop, double-buffered fragments ---------------
  v16h a0A, a1A, bA[NT];          // buffer A
  v16h a0B, a1B, bB[NT];          // buffer B
  a0A = load_afrag(ap0);
  a1A = load_afrag(ap1);
#pragma unroll
  for (int ni = 0; ni < NT; ++ni) bA[ni] = load_bfrag(bp[ni]);

  int k = 0;
  for (; k + 64 <= g.K; k += 64) {
    if (k + 128 < g.K) {
      __builtin_prefetch(ap0 + k + 128, 0, 3);
      __builtin_prefetch(ap1 + k + 128, 0, 3);
#pragma unroll
      for (int ni = 0; ni < NT; ++ni) __builtin_prefetch(bp[ni] + k + 128, 0, 3);
    }
    // load buffer B for step k+32, compute with buffer A (step k)
    a0B = load_afrag(ap0 + k + 32);
    a1B = load_afrag(ap1 + k + 32);
#pragma unroll
    for (int ni = 0; ni < NT; ++ni) bB[ni] = load_bfrag(bp[ni] + k + 32);
#pragma unroll
    for (int ni = 0; ni < NT; ++ni) {
      acc[0][ni] = wmma_step(a0A, bA[ni], acc[0][ni]);
      acc[1][ni] = wmma_step(a1A, bA[ni], acc[1][ni]);
    }
    // load buffer A for step k+64 (if any), compute with buffer B (k+32)
    if (k + 64 < g.K) {
      a0A = load_afrag(ap0 + k + 64);
      a1A = load_afrag(ap1 + k + 64);
#pragma unroll
      for (int ni = 0; ni < NT; ++ni) bA[ni] = load_bfrag(bp[ni] + k + 64);
    }
#pragma unroll
    for (int ni = 0; ni < NT; ++ni) {
      acc[0][ni] = wmma_step(a0B, bB[ni], acc[0][ni]);
      acc[1][ni] = wmma_step(a1B, bB[ni], acc[1][ni]);
    }
  }
  if (k < g.K) {                  // odd tail step (fragments already in buf A)
#pragma unroll
    for (int ni = 0; ni < NT; ++ni) {
      acc[0][ni] = wmma_step(a0A, bA[ni], acc[0][ni]);
      acc[1][ni] = wmma_step(a1A, bA[ni], acc[1][ni]);
    }
  }

  // D layout: VGPR r, lane<16 -> (M=r, N=lane); lane>=16 -> (M=r+8, N=lane-16)
  const int cc  = lane & 15;
  const int rb8 = (lane >> 4) * 8;
#pragma unroll
  for (int ni = 0; ni < NT; ++ni) {
    const int ncol = n0 + ni * 16 + cc;
    float bv = 0.0f;
    if constexpr (BIAS) bv = g.bias[ncol];
#pragma unroll
    for (int mi = 0; mi < 2; ++mi) {
#pragma unroll
      for (int r = 0; r < 8; ++r) {
        float v = acc[mi][ni][r] + bv;
        if constexpr (RELU) v = fmaxf(v, 0.0f);
        long long m = m0 + mi * 16 + rb8 + r;
        C[m * (long long)g.rsC + (long long)ncol * g.csC] = (TC)v;
      }
    }
  }
}

// ---------------- embedding: out[t,e] = emb[tok[t],e] + pos[t%S,e] ----------
__global__ __launch_bounds__(256) void k_embed(const int* __restrict__ tok,
                                               const float* __restrict__ emb,
                                               const float* __restrict__ pos,
                                               float* __restrict__ out) {
  const int t = blockIdx.x, e = threadIdx.x;
  const int s = t % TS;
  const long long id = tok[t];
  out[(long long)t * TE + e] = emb[id * TE + e] + pos[(long long)s * TE + e];
}

// ---------------- masked softmax over K=1024 (row per block) ----------------
// mode 0: key-padding mask from src tokens; mode 1: causal.
__global__ __launch_bounds__(256) void k_softmax(const float* __restrict__ sim,
                                                 _Float16* __restrict__ att,
                                                 const int* __restrict__ mtok,
                                                 int mode, float invscale) {
  const long long row = blockIdx.x;           // row = (n*H+h)*S + q
  const int q  = (int)(row % TS);
  const int bh = (int)(row / TS);
  const int n  = bh / TH;
  const float* srcp = sim + row * TS;
  _Float16*    dstp = att + row * TS;
  const int t = threadIdx.x;

  float v[4]; bool ok[4];
  float mx = -3.4e38f;
#pragma unroll
  for (int j = 0; j < 4; ++j) {
    int k = t + j * 256;
    bool m = (mode == 0) ? (mtok[(long long)n * TS + k] != 0) : (k <= q);
    ok[j] = m;
    v[j]  = m ? srcp[k] * invscale : -3.4e38f;
    mx    = fmaxf(mx, v[j]);
  }
  __shared__ float red[256];
  red[t] = mx; __syncthreads();
  for (int o = 128; o > 0; o >>= 1) { if (t < o) red[t] = fmaxf(red[t], red[t + o]); __syncthreads(); }
  mx = red[0]; __syncthreads();

  float e[4]; float sum = 0.0f;
#pragma unroll
  for (int j = 0; j < 4; ++j) { e[j] = ok[j] ? __expf(v[j] - mx) : 0.0f; sum += e[j]; }
  red[t] = sum; __syncthreads();
  for (int o = 128; o > 0; o >>= 1) { if (t < o) red[t] += red[t + o]; __syncthreads(); }
  sum = red[0];
  const float inv = (sum > 0.0f) ? 1.0f / sum : 0.0f;
#pragma unroll
  for (int j = 0; j < 4; ++j) dstp[t + j * 256] = (_Float16)(e[j] * inv);
}

// ---------------- out = LayerNorm(a + b) * g + beta  (E=256 per block) ------
__global__ __launch_bounds__(256) void k_addln(const float* __restrict__ a,
                                               const float* __restrict__ b,
                                               const float* __restrict__ gg,
                                               const float* __restrict__ bb,
                                               float* __restrict__ out) {
  const int t = threadIdx.x;
  const long long base = (long long)blockIdx.x * TE;
  float x = a[base + t] + b[base + t];
  __shared__ float red[256];
  red[t] = x; __syncthreads();
  for (int o = 128; o > 0; o >>= 1) { if (t < o) red[t] += red[t + o]; __syncthreads(); }
  const float mean = red[0] * (1.0f / TE); __syncthreads();
  const float d = x - mean;
  red[t] = d * d; __syncthreads();
  for (int o = 128; o > 0; o >>= 1) { if (t < o) red[t] += red[t + o]; __syncthreads(); }
  const float var = red[0] * (1.0f / TE);
  out[base + t] = d * rsqrtf(var + 1e-5f) * gg[t] + bb[t];
}

// ---------------------------------------------------------------------------
extern "C" void kernel_launch(void* const* d_in, const int* in_sizes, int n_in,
                              void* d_out, int out_size, void* d_ws, size_t ws_size,
                              hipStream_t stream) {
  (void)in_sizes; (void)n_in; (void)out_size; (void)ws_size;

  const int S = TS, E = TE, H = TH, D = TD, FE = TFE;
  const int Ntok = TNTOK, NH = TNH;
  const float INVSCALE = 1.0f / 16.0f;   // 1/sqrt(E)

  // ---- inputs (recursive dict-insertion-order flattening of setup_inputs) --
  const int*   src      = (const int*)d_in[0];
  const int*   trg      = (const int*)d_in[1];
  const float* enc_emb  = (const float*)d_in[2];
  const float* enc_pos  = (const float*)d_in[3];
  const float* dec_emb  = (const float*)d_in[4];
  const float* dec_pos  = (const float*)d_in[5];
  // enc block: Wv Wq Wk Wo bo g1 b1 g2 b2 W1 c1 W2 c2  -> 6..18
  const float* eWv = (const float*)d_in[6];
  const float* eWq = (const float*)d_in[7];
  /* d_in[8] = enc.Wk, dead (faithful to source) */
  const float* eWo = (const float*)d_in[9];
  const float* ebo = (const float*)d_in[10];
  const float* eg1 = (const float*)d_in[11];
  const float* eb1 = (const float*)d_in[12];
  const float* eg2 = (const float*)d_in[13];
  const float* eb2 = (const float*)d_in[14];
  const float* eW1 = (const float*)d_in[15];
  const float* ec1 = (const float*)d_in[16];
  const float* eW2 = (const float*)d_in[17];
  const float* ec2 = (const float*)d_in[18];
  // dec_self: Wv Wq Wk Wo bo ng nb -> 19..25
  const float* sWv = (const float*)d_in[19];
  const float* sWq = (const float*)d_in[20];
  /* d_in[21] = dec_self.Wk, dead */
  const float* sWo = (const float*)d_in[22];
  const float* sbo = (const float*)d_in[23];
  const float* sng = (const float*)d_in[24];
  const float* snb = (const float*)d_in[25];
  // dec_cross: Wv Wq Wk Wo bo g1 b1 g2 b2 W1 c1 W2 c2 -> 26..38
  const float* cWv = (const float*)d_in[26];
  const float* cWq = (const float*)d_in[27];
  /* d_in[28] = dec_cross.Wk, dead */
  const float* cWo = (const float*)d_in[29];
  const float* cbo = (const float*)d_in[30];
  const float* cg1 = (const float*)d_in[31];
  const float* cb1 = (const float*)d_in[32];
  const float* cg2 = (const float*)d_in[33];
  const float* cb2 = (const float*)d_in[34];
  const float* cW1 = (const float*)d_in[35];
  const float* cc1 = (const float*)d_in[36];
  const float* cW2 = (const float*)d_in[37];
  const float* cc2 = (const float*)d_in[38];
  const float* out_W = (const float*)d_in[39];
  const float* out_b = (const float*)d_in[40];
  float* logits = (float*)d_out;

  // ---- workspace carve (~124 MB) -------------------------------------------
  char* w = (char*)d_ws;
  auto carve = [&](size_t bytes) -> void* {
    void* p = (void*)w;
    w += (bytes + 255) & ~(size_t)255;
    return p;
  };
  float*     x      = (float*)carve((size_t)Ntok * E * 4);        // enc act
  float*     y      = (float*)carve((size_t)Ntok * E * 4);        // dec act
  float*     qln    = (float*)carve((size_t)Ntok * E * 4);
  float*     attout = (float*)carve((size_t)Ntok * E * 4);
  float*     obuf   = (float*)carve((size_t)Ntok * E * 4);
  float*     rbuf   = (float*)carve((size_t)Ntok * E * 4);
  float*     ff1    = (float*)carve((size_t)Ntok * FE * E * 4);
  float*     ff2    = (float*)carve((size_t)Ntok * E * 4);
  _Float16*  qh     = (_Float16*)carve((size_t)Ntok * E * 2);     // [tok, H*D]
  _Float16*  vhT    = (_Float16*)carve((size_t)H * D * Ntok * 2); // [h, d, tok]
  float*     sim    = (float*)carve((size_t)NH * S * S * 4);      // [n*H+h, q, k]
  _Float16*  att    = (_Float16*)carve((size_t)NH * S * S * 2);

  // ---- GEMM launchers ------------------------------------------------------
  auto projQ = [&](const float* in, const float* W) {  // qh[t, h*D+e] f16
    GArgs g{}; g.A = in;  g.lda = E;   g.sAi = D;  g.sAo = 0;
    g.B = W;   g.ldb = D;
    g.C = qh;  g.rsC = E; g.csC = 1;   g.sCi = D;  g.sCo = 0;
    g.M = Ntok; g.Nc = D; g.K = D; g.bi = H; g.nb = H;
    dim3 gr(1, (Ntok + 127) / 128, H);
    gemm_wmma<float, float, _Float16, false, false, 2><<<gr, 128, 0, stream>>>(g);
  };
  auto projV = [&](const float* in, const float* W) {  // vhT[h, d, t] f16
    GArgs g{}; g.A = in;  g.lda = E;    g.sAi = D;              g.sAo = 0;
    g.B = W;   g.ldb = D;
    g.C = vhT; g.rsC = 1; g.csC = Ntok; g.sCi = (long long)D * Ntok; g.sCo = 0;
    g.M = Ntok; g.Nc = D; g.K = D; g.bi = H; g.nb = H;
    dim3 gr(1, (Ntok + 127) / 128, H);
    gemm_wmma<float, float, _Float16, false, false, 2><<<gr, 128, 0, stream>>>(g);
  };
  auto qkgemm = [&](const float* keys) {               // sim = qh @ keys^T
    GArgs g{}; g.A = qh;  g.lda = E; g.sAi = D; g.sAo = (long long)S * E;
    g.B = keys; g.ldb = E; g.sBi = D; g.sBo = (long long)S * E;
    g.C = sim;  g.rsC = S; g.csC = 1;
    g.sCi = (long long)S * S; g.sCo = (long long)H * S * S;
    g.M = S; g.Nc = S; g.K = D; g.bi = H; g.nb = NH;
    dim3 gr(S / 64, (S + 127) / 128, NH);
    gemm_wmma<_Float16, float, float, false, false, 4><<<gr, 128, 0, stream>>>(g);
  };
  auto avgemm = [&]() {                                // attout = att @ v
    GArgs g{}; g.A = att; g.lda = S;
    g.sAi = (long long)S * S; g.sAo = (long long)H * S * S;
    g.B = vhT; g.ldb = Ntok; g.sBi = (long long)D * Ntok; g.sBo = S;
    g.C = attout; g.rsC = E; g.csC = 1; g.sCi = D; g.sCo = (long long)S * E;
    g.M = S; g.Nc = D; g.K = S; g.bi = H; g.nb = NH;
    dim3 gr(1, (S + 127) / 128, NH);
    gemm_wmma<_Float16, _Float16, float, false, false, 2><<<gr, 128, 0, stream>>>(g);
  };
  auto dense = [&](const float* Ain, const float* W, const float* bias,
                   float* Cout, int M, int Nc, int K, bool relu) {
    GArgs g{}; g.A = Ain; g.lda = K; g.B = W; g.ldb = K; g.bias = bias;
    g.C = Cout; g.rsC = Nc; g.csC = 1; g.M = M; g.Nc = Nc; g.K = K;
    g.bi = 1; g.nb = 1;
    dim3 gr(Nc / 64, (M + 127) / 128, 1);
    if (relu) gemm_wmma<float, float, float, true, true, 4><<<gr, 128, 0, stream>>>(g);
    else      gemm_wmma<float, float, float, true, false, 4><<<gr, 128, 0, stream>>>(g);
  };
  auto softmax = [&](int mode, const int* mtokens) {
    k_softmax<<<NH * S, 256, 0, stream>>>(sim, att, mtokens, mode, INVSCALE);
  };
  auto addln = [&](const float* a, const float* b, const float* gg,
                   const float* bb, float* outp) {
    k_addln<<<Ntok, 256, 0, stream>>>(a, b, gg, bb, outp);
  };

  // ---- encoder -------------------------------------------------------------
  k_embed<<<Ntok, 256, 0, stream>>>(src, enc_emb, enc_pos, x);
  for (int i = 0; i < TL; ++i) {
    projQ(x, eWq + (long long)i * D * D);
    projV(x, eWv + (long long)i * D * D);
    qkgemm(x);                               // keys = x (unprojected, faithful)
    softmax(0, src);
    avgemm();
    dense(attout, eWo + (long long)i * E * E, ebo + (long long)i * E, obuf, Ntok, E, E, false);
    addln(obuf, x, eg1 + (long long)i * E, eb1 + (long long)i * E, rbuf);
    dense(rbuf, eW1 + (long long)i * FE * E * E, ec1 + (long long)i * FE * E, ff1, Ntok, FE * E, E, true);
    dense(ff1, eW2 + (long long)i * E * FE * E, ec2 + (long long)i * E, ff2, Ntok, E, FE * E, false);
    addln(rbuf, ff2, eg2 + (long long)i * E, eb2 + (long long)i * E, x);
  }

  // ---- decoder -------------------------------------------------------------
  k_embed<<<Ntok, 256, 0, stream>>>(trg, dec_emb, dec_pos, y);
  for (int i = 0; i < TL; ++i) {
    // self attention (causal)
    projQ(y, sWq + (long long)i * D * D);
    projV(y, sWv + (long long)i * D * D);
    qkgemm(y);
    softmax(1, src);
    avgemm();
    dense(attout, sWo + (long long)i * E * E, sbo + (long long)i * E, obuf, Ntok, E, E, false);
    addln(obuf, y, sng + (long long)i * E, snb + (long long)i * E, qln);
    // cross attention (v = k = encoder x, q = qln)
    projQ(qln, cWq + (long long)i * D * D);
    projV(x,   cWv + (long long)i * D * D);
    qkgemm(x);
    softmax(0, src);
    avgemm();
    dense(attout, cWo + (long long)i * E * E, cbo + (long long)i * E, obuf, Ntok, E, E, false);
    addln(obuf, qln, cg1 + (long long)i * E, cb1 + (long long)i * E, rbuf);
    dense(rbuf, cW1 + (long long)i * FE * E * E, cc1 + (long long)i * FE * E, ff1, Ntok, FE * E, E, true);
    dense(ff1, cW2 + (long long)i * E * FE * E, cc2 + (long long)i * E, ff2, Ntok, E, FE * E, false);
    addln(rbuf, ff2, cg2 + (long long)i * E, cb2 + (long long)i * E, y);
  }

  // ---- output projection: logits = y @ out_W^T + out_b ---------------------
  dense(y, out_W, out_b, logits, Ntok, 32000, E, false);
}
